// Adaptive_90005334655191
// MI455X (gfx1250) — matrix operand order repaired and checked by reading, not compiled, not run on hardware
//
#include <hip/hip_runtime.h>
#include <hip/hip_bf16.h>

// ---- shapes (compile-time; harness problem is fixed-size) ----
#define BB   128
#define TT   64
#define HH   512
#define EE   1024
#define VOCN 10000
#define KK   49
#define MM   (BB * TT)          // 8192 rows
#define NT_OUT (VOCN / 16)      // 625 N tiles in final GEMM
#define MB   4                  // M-register-blocking: 4 x 16 = 64 rows per wave

typedef __attribute__((ext_vector_type(16))) __bf16 v16bf;
typedef __attribute__((ext_vector_type(8)))  float  v8f;
typedef unsigned short u16;

// round-to-nearest-even f32 -> bf16
static __device__ __forceinline__ u16 f2bf(float f) {
  unsigned int u = __float_as_uint(f);
  u += 0x7FFFu + ((u >> 16) & 1u);
  return (u16)(u >> 16);
}

// ---------------- packing kernels ----------------
__global__ void f32_to_bf16_kernel(const float* __restrict__ src,
                                   u16* __restrict__ dst, int n) {
  int i = blockIdx.x * blockDim.x + threadIdx.x;
  if (i < n) dst[i] = f2bf(src[i]);
}

// hiddens_{t-1}: per batch, shift one timestep, t==0 -> 0
__global__ void shift_bf16_kernel(const float* __restrict__ hiddens,
                                  u16* __restrict__ dst, int n) {
  int i = blockIdx.x * blockDim.x + threadIdx.x;
  if (i >= n) return;
  int r = i % (TT * HH);
  int t = r / HH;
  dst[i] = (t == 0) ? (u16)0 : f2bf(hiddens[i - HH]);
}

// ---------------- WMMA fragment loaders (ISA 7.12.2, 16-bit, wave32) ----
// A: 16x32 bf16, row-major source A[m][k], lda in elements.
//   lane L: M = L&15, half = L>>4 ; VGPR v: K = (v>=4?16:0) + 8*half + (v&3)*2 (+pos)
static __device__ __forceinline__ v16bf load_a_frag(const u16* __restrict__ A,
                                                    int lda, int m0, int k0, int lane) {
  const int M = lane & 15, half = lane >> 4;
  const u16* row = A + (size_t)(m0 + M) * lda + k0;
  union { v16bf v; unsigned int u[8]; } r;
#pragma unroll
  for (int v2 = 0; v2 < 8; ++v2) {
    const int kb = ((v2 >= 4) ? 16 : 0) + half * 8 + (v2 & 3) * 2;
    r.u[v2] = *(const unsigned int*)(row + kb);
  }
  return r.v;
}

// B: 32x16 bf16 with B[k][n] = W[n][k], W row-major NxK (ldw in elements).
//   lane L: N = L&15, half = L>>4 ; VGPR v: K = (v>=4?16:0) + 8*half + (v&3)*2 (+pos)
static __device__ __forceinline__ v16bf load_b_frag(const u16* __restrict__ W,
                                                    int ldw, int n0, int k0, int lane) {
  const int N = lane & 15, half = lane >> 4;
  const u16* row = W + (size_t)(n0 + N) * ldw + k0;
  union { v16bf v; unsigned int u[8]; } r;
#pragma unroll
  for (int v2 = 0; v2 < 8; ++v2) r.u[v2] = *(const unsigned int*)(row + ((v2 >= 4) ? 16 : 0) + half * 8 + (v2 & 3) * 2);
  return r.v;
}

// ---------------- gate GEMM + sigmoid*tanh epilogue ----------------
// gate[m][h] = sum_e x[m][e]*Wx[h][e] + sum_j hm1[m][j]*Wsh[h][j]
// sent = sigmoid(gate) * tanh(cells)
// Each wave: 64x16 strip (4 acc tiles share one B fragment per k-step).
// waves = (8192/64) * (512/16) = 4096  -> 512 blocks x 8 waves
__global__ __launch_bounds__(256) void gate_sent_kernel(
    const u16* __restrict__ xbf,  const u16* __restrict__ hm1bf,
    const u16* __restrict__ Wxbf, const u16* __restrict__ Wshbf,
    const float* __restrict__ cells, float* __restrict__ sent) {
  const int w    = blockIdx.x * 8 + (threadIdx.x >> 5);
  const int lane = threadIdx.x & 31;
  const int m0 = (w >> 5) * (16 * MB);  // 128 M-blocks
  const int n0 = (w & 31) * 16;         // 32  H-tiles
  v8f acc[MB] = {};
  for (int k0 = 0; k0 < EE; k0 += 32) {
    const v16bf bfrag = load_b_frag(Wxbf, EE, n0, k0, lane);
#pragma unroll
    for (int i = 0; i < MB; ++i)
      acc[i] = __builtin_amdgcn_wmma_f32_16x16x32_bf16(
          false, load_a_frag(xbf, EE, m0 + 16 * i, k0, lane),
          false, bfrag, (short)0, acc[i], false, false);
  }
  for (int k0 = 0; k0 < HH; k0 += 32) {
    const v16bf bfrag = load_b_frag(Wshbf, HH, n0, k0, lane);
#pragma unroll
    for (int i = 0; i < MB; ++i)
      acc[i] = __builtin_amdgcn_wmma_f32_16x16x32_bf16(
          false, load_a_frag(hm1bf, HH, m0 + 16 * i, k0, lane),
          false, bfrag, (short)0, acc[i], false, false);
  }
  const int half = lane >> 4;
  const int ncol = n0 + (lane & 15);
#pragma unroll
  for (int i = 0; i < MB; ++i) {
#pragma unroll
    for (int r = 0; r < 8; ++r) {
      const int m = m0 + 16 * i + r + 8 * half;
      const float g = acc[i][r];
      const float s = (1.0f / (1.0f + expf(-g))) * tanhf(cells[(size_t)m * HH + ncol]);
      sent[(size_t)m * HH + ncol] = s;
    }
  }
}

// ---------------- cv[b][k][n] = sum_h V[b][k][h] * Wv[n][h]  (f32 exact) ----
__global__ void cv_kernel(const float* __restrict__ V, const float* __restrict__ Wv,
                          float* __restrict__ cv) {
  int idx = blockIdx.x * blockDim.x + threadIdx.x;
  if (idx >= BB * KK * KK) return;
  const int n = idx % KK, k = (idx / KK) % KK, b = idx / (KK * KK);
  const float4* vp = (const float4*)(V + ((size_t)b * KK + k) * HH);
  const float4* wp = (const float4*)(Wv + (size_t)n * HH);
  float acc = 0.f;
#pragma unroll 4
  for (int i = 0; i < HH / 4; ++i) {
    float4 a = vp[i], w = wp[i];
    acc += a.x * w.x + a.y * w.y + a.z * w.z + a.w * w.w;
  }
  cv[idx] = acc;
}

// ---- cg = hiddens@Wg^T ; cs = sent@Ws^T + cg   (f32 exact, K=49 columns) ---
__global__ void cgcs_kernel(const float* __restrict__ hiddens,
                            const float* __restrict__ sent,
                            const float* __restrict__ Wg,
                            const float* __restrict__ Ws,
                            float* __restrict__ cg, float* __restrict__ cs) {
  int idx = blockIdx.x * blockDim.x + threadIdx.x;
  if (idx >= MM * KK) return;
  const int n = idx % KK, m = idx / KK;
  const float4* hp = (const float4*)(hiddens + (size_t)m * HH);
  const float4* sp = (const float4*)(sent + (size_t)m * HH);
  const float4* gp = (const float4*)(Wg + (size_t)n * HH);
  const float4* wp = (const float4*)(Ws + (size_t)n * HH);
  float a = 0.f, c = 0.f;
#pragma unroll 4
  for (int i = 0; i < HH / 4; ++i) {
    float4 h = hp[i], s = sp[i], g = gp[i], w = wp[i];
    a += h.x * g.x + h.y * g.y + h.z * g.z + h.w * g.w;
    c += s.x * w.x + s.y * w.y + s.z * w.z + s.w * w.w;
  }
  cg[idx] = a;
  cs[idx] = c + a;
}

// --------- attention: z, alpha, beta, c, c_hat, pack A=(c_hat+hiddens) ------
// one block of 64 threads per (b,t)
__global__ __launch_bounds__(64) void attn_kernel(
    const float* __restrict__ cv, const float* __restrict__ cg,
    const float* __restrict__ cs, const float* __restrict__ wh,
    const float* __restrict__ V,  const float* __restrict__ sent,
    const float* __restrict__ hiddens, u16* __restrict__ Abf,
    float* __restrict__ alpha_out, float* __restrict__ beta_out) {
  const int m = blockIdx.x;        // b*T + t
  const int b = m / TT;
  const int tid = threadIdx.x;
  __shared__ float zsh[KK];
  __shared__ float alsh[KK];
  __shared__ float bsh;

  const float* cgm = cg + (size_t)m * KK;
  if (tid < KK) {
    const float* cvk = cv + ((size_t)b * KK + tid) * KK;
    float acc = 0.f;
    for (int n = 0; n < KK; ++n) acc += tanhf(cvk[n] + cgm[n]) * wh[n];
    zsh[tid] = acc;
  } else if (tid == KK) {
    const float* csm = cs + (size_t)m * KK;
    float acc = 0.f;
    for (int n = 0; n < KK; ++n) acc += tanhf(csm[n]) * wh[n];
    bsh = acc;                                  // z_ext (temp)
  }
  __syncthreads();

  if (tid == 0) {
    const float ze = bsh;
    float mz = -1e30f;
    for (int k = 0; k < KK; ++k) mz = fmaxf(mz, zsh[k]);
    float s = 0.f;
    for (int k = 0; k < KK; ++k) { float e = expf(zsh[k] - mz); alsh[k] = e; s += e; }
    const float inv = 1.f / s;
    for (int k = 0; k < KK; ++k) {
      alsh[k] *= inv;
      alpha_out[(size_t)m * KK + k] = alsh[k];
    }
    const float me = fmaxf(mz, ze);
    const float se = expf(ze - me);
    const float beta = se / (s * expf(mz - me) + se);
    bsh = beta;
    beta_out[m] = beta;
  }
  __syncthreads();

  const float beta = bsh;
  const float* Vb = V + (size_t)b * KK * HH;
  for (int h = tid; h < HH; h += 64) {
    float c = 0.f;
#pragma unroll 7
    for (int k = 0; k < KK; ++k) c += alsh[k] * Vb[(size_t)k * HH + h];
    const float ch = beta * sent[(size_t)m * HH + h] + (1.f - beta) * c;
    Abf[(size_t)m * HH + h] = f2bf(ch + hiddens[(size_t)m * HH + h]);
  }
}

// ---------------- final GEMM: out = A @ Wm^T + bm  (8192 x 10000 x 512) ----
// Each wave: 64x16 strip (4 acc tiles share one B fragment per k-step).
// waves = (8192/64) * 625 = 80000  -> 10000 blocks x 8 waves
__global__ __launch_bounds__(256) void out_gemm_kernel(
    const u16* __restrict__ Abf, const u16* __restrict__ Wmbf,
    const float* __restrict__ bm, float* __restrict__ out) {
  const int w    = blockIdx.x * 8 + (threadIdx.x >> 5);
  const int lane = threadIdx.x & 31;
  const int m0 = (w / NT_OUT) * (16 * MB);
  const int n0 = (w % NT_OUT) * 16;
  v8f acc[MB] = {};
  for (int k0 = 0; k0 < HH; k0 += 32) {
    if (k0 + 32 < HH) {   // prefetch next K panel (-> global_prefetch_b8)
      __builtin_prefetch(Abf  + (size_t)(m0 + lane) * HH + k0 + 32, 0, 1);
      __builtin_prefetch(Abf  + (size_t)(m0 + 32 + lane) * HH + k0 + 32, 0, 1);
      __builtin_prefetch(Wmbf + (size_t)(n0 + (lane & 15)) * HH + k0 + 32, 0, 1);
    }
    const v16bf bfrag = load_b_frag(Wmbf, HH, n0, k0, lane);
#pragma unroll
    for (int i = 0; i < MB; ++i)
      acc[i] = __builtin_amdgcn_wmma_f32_16x16x32_bf16(
          false, load_a_frag(Abf, HH, m0 + 16 * i, k0, lane),
          false, bfrag, (short)0, acc[i], false, false);
  }
  const int half = lane >> 4;
  const int ncol = n0 + (lane & 15);
  const float bias = bm[ncol];
#pragma unroll
  for (int i = 0; i < MB; ++i) {
#pragma unroll
    for (int r = 0; r < 8; ++r) {
      out[(size_t)(m0 + 16 * i + r + 8 * half) * VOCN + ncol] = acc[i][r] + bias;
    }
  }
}

// ---------------- host launcher ----------------
extern "C" void kernel_launch(void* const* d_in, const int* in_sizes, int n_in,
                              void* d_out, int out_size, void* d_ws, size_t ws_size,
                              hipStream_t stream) {
  const float* x       = (const float*)d_in[0];
  const float* hiddens = (const float*)d_in[1];
  const float* cells   = (const float*)d_in[2];
  const float* V       = (const float*)d_in[3];
  const float* Wv      = (const float*)d_in[4];
  const float* Wg      = (const float*)d_in[5];
  const float* Ws      = (const float*)d_in[6];
  const float* wh      = (const float*)d_in[7];
  const float* Wx      = (const float*)d_in[8];
  const float* Wsh     = (const float*)d_in[9];
  const float* Wm      = (const float*)d_in[10];
  const float* bm      = (const float*)d_in[11];

  char* ws = (char*)d_ws;
  size_t off = 0;
  auto carve = [&](size_t bytes) -> char* {
    char* p = ws + off;
    off += (bytes + 255) & ~(size_t)255;
    return p;
  };
  u16*   xbf   = (u16*)  carve((size_t)MM * EE * 2);
  u16*   hm1bf = (u16*)  carve((size_t)MM * HH * 2);
  u16*   Wxbf  = (u16*)  carve((size_t)HH * EE * 2);
  u16*   Wshbf = (u16*)  carve((size_t)HH * HH * 2);
  u16*   Wmbf  = (u16*)  carve((size_t)VOCN * HH * 2);
  float* sent  = (float*)carve((size_t)MM * HH * 4);
  float* cv    = (float*)carve((size_t)BB * KK * KK * 4);
  float* cg    = (float*)carve((size_t)MM * KK * 4);
  float* cs    = (float*)carve((size_t)MM * KK * 4);
  u16*   Abf   = (u16*)  carve((size_t)MM * HH * 2);

  float* out_p   = (float*)d_out;                      // (MM, VOC)
  float* alpha_p = out_p + (size_t)MM * VOCN;          // (MM, KK)
  float* beta_p  = alpha_p + (size_t)MM * KK;          // (MM,)

  const int thr = 256;
  f32_to_bf16_kernel<<<(MM * EE + thr - 1) / thr, thr, 0, stream>>>(x, xbf, MM * EE);
  f32_to_bf16_kernel<<<(HH * EE + thr - 1) / thr, thr, 0, stream>>>(Wx, Wxbf, HH * EE);
  f32_to_bf16_kernel<<<(HH * HH + thr - 1) / thr, thr, 0, stream>>>(Wsh, Wshbf, HH * HH);
  f32_to_bf16_kernel<<<(VOCN * HH + thr - 1) / thr, thr, 0, stream>>>(Wm, Wmbf, VOCN * HH);
  shift_bf16_kernel<<<(MM * HH + thr - 1) / thr, thr, 0, stream>>>(hiddens, hm1bf, MM * HH);

  // waves = (MM/64) * (HH/16) = 4096 -> 512 blocks
  gate_sent_kernel<<<(MM / (16 * MB)) * (HH / 16) / 8, 256, 0, stream>>>(
      xbf, hm1bf, Wxbf, Wshbf, cells, sent);

  cv_kernel<<<(BB * KK * KK + thr - 1) / thr, thr, 0, stream>>>(V, Wv, cv);
  cgcs_kernel<<<(MM * KK + thr - 1) / thr, thr, 0, stream>>>(hiddens, sent, Wg, Ws, cg, cs);

  attn_kernel<<<MM, 64, 0, stream>>>(cv, cg, cs, wh, V, sent, hiddens, Abf,
                                     alpha_p, beta_p);

  // waves = (MM/64) * 625 = 80000 -> 10000 blocks
  out_gemm_kernel<<<(MM / (16 * MB)) * NT_OUT / 8, 256, 0, stream>>>(Abf, Wmbf, bm, out_p);
}
